// NaiveMPNN_28613072126231
// MI455X (gfx1250) — matrix operand order repaired and checked by reading, not compiled, not run on hardware
//
#include <hip/hip_runtime.h>
#include <hip/hip_bf16.h>

// ---------------------------------------------------------------------------
// Problem constants (from reference)
// ---------------------------------------------------------------------------
#define NV_   4096   // V
#define NE_   8192   // E
#define HV_   256
#define HE_   128
#define OUT_  256
#define EPS_  1e-6f

typedef __bf16 v16bf __attribute__((ext_vector_type(16)));
typedef __bf16 v8bf  __attribute__((ext_vector_type(8)));
typedef float  v8f   __attribute__((ext_vector_type(8)));

// ---------------------------------------------------------------------------
// 1) Recover edge endpoint indices from dense one-hot incidence [V, E].
// ---------------------------------------------------------------------------
__global__ void k_extract_idx(const float4* __restrict__ vew, int* __restrict__ idx,
                              int total4) {
    int t = blockIdx.x * blockDim.x + threadIdx.x;
    if (t >= total4) return;
    float4 val = vew[t];
    int base = t << 2;
    int v = base >> 13;        // / NE_
    int e = base & (NE_ - 1);  // % NE_
    if (val.x > 0.5f) idx[e + 0] = v;
    if (val.y > 0.5f) idx[e + 1] = v;
    if (val.z > 0.5f) idx[e + 2] = v;
    if (val.w > 0.5f) idx[e + 3] = v;
}

// ---------------------------------------------------------------------------
// 2a) a[v] = hv[v,:]·w_att[0:256], c[v] = hv[v,:]·w_att[384:640]; wave per row
// ---------------------------------------------------------------------------
__global__ void k_att_hv(const float* __restrict__ hv, const float* __restrict__ w_att,
                         float* __restrict__ a_vec, float* __restrict__ c_vec) {
    int wid  = (blockIdx.x * blockDim.x + threadIdx.x) >> 5;
    int lane = threadIdx.x & 31;
    if (wid >= NV_) return;
    const float* row = hv + (size_t)wid * HV_;
    float sa = 0.f, sc = 0.f;
    #pragma unroll
    for (int i = lane; i < HV_; i += 32) {
        float x = row[i];
        sa += x * w_att[i];
        sc += x * w_att[HV_ + HE_ + i];
    }
    #pragma unroll
    for (int off = 16; off > 0; off >>= 1) {
        sa += __shfl_down(sa, off, 32);
        sc += __shfl_down(sc, off, 32);
    }
    if (lane == 0) { a_vec[wid] = sa; c_vec[wid] = sc; }
}

// 2b) be[e] = he[e,:]·w_att[256:384]; wave per row
__global__ void k_att_he(const float* __restrict__ he, const float* __restrict__ w_att,
                         float* __restrict__ be_vec) {
    int wid  = (blockIdx.x * blockDim.x + threadIdx.x) >> 5;
    int lane = threadIdx.x & 31;
    if (wid >= NE_) return;
    const float* row = he + (size_t)wid * HE_;
    float s = 0.f;
    #pragma unroll
    for (int i = lane; i < HE_; i += 32) s += row[i] * w_att[HV_ + i];
    #pragma unroll
    for (int off = 16; off > 0; off >>= 1) s += __shfl_down(s, off, 32);
    if (lane == 0) be_vec[wid] = s;
}

// ---------------------------------------------------------------------------
// 3) Per (directed) edge j: edge_w = sigmoid(a[u]+be+c[v]+b_att) + degree sums
// ---------------------------------------------------------------------------
__global__ void k_edge(const int* __restrict__ u_idx, const int* __restrict__ v_idx,
                       const float* __restrict__ a_vec, const float* __restrict__ c_vec,
                       const float* __restrict__ be_vec, const float* __restrict__ b_att,
                       float* __restrict__ edge_w,
                       float* __restrict__ r_sum, float* __restrict__ c_sum) {
    int j = blockIdx.x * blockDim.x + threadIdx.x;
    if (j >= 2 * NE_) return;
    int e = (j < NE_) ? j : j - NE_;
    int u = (j < NE_) ? u_idx[e] : v_idx[e];
    int v = (j < NE_) ? v_idx[e] : u_idx[e];
    float z = a_vec[u] + be_vec[e] + c_vec[v] + b_att[0];
    float w = 1.f / (1.f + __expf(-z));
    edge_w[j] = w;
    atomicAdd(&r_sum[u], w);
    atomicAdd(&c_sum[v], w);
}

// ---------------------------------------------------------------------------
// 4a) hv_ftr -> first HV_ columns of the [V, 2*HV] concat buffer (bf16)
// ---------------------------------------------------------------------------
__global__ void k_hv2cat(const float* __restrict__ hv, __bf16* __restrict__ h_cat) {
    int t = blockIdx.x * blockDim.x + threadIdx.x;
    if (t >= NV_ * HV_) return;
    int v = t >> 8, h = t & (HV_ - 1);
    h_cat[(size_t)v * (2 * HV_) + h] = (__bf16)hv[t];
}

// ---------------------------------------------------------------------------
// 4b) Pack weight B[K,N] (f32 row-major) into WMMA fragment order (bf16):
//     Bp[((kt*ntn + nt)*32 + lane)*16 + i] = B[kt*32 + (i&7) + 8*(lane>>4) + 16*(i>>3)][nt*16 + (lane&15)]
//     One wave's B fragment then loads as one contiguous 32B/lane chunk.
// ---------------------------------------------------------------------------
__global__ void k_packB(const float* __restrict__ B, __bf16* __restrict__ Bp,
                        int K, int N) {
    int t = blockIdx.x * blockDim.x + threadIdx.x;
    if (t >= K * N) return;
    int i    = t & 15;
    int lane = (t >> 4) & 31;
    int rest = t >> 9;            // (kt, nt) pair index
    int ntn  = N >> 4;
    int nt   = rest % ntn;
    int kt   = rest / ntn;
    int kb   = kt * 32 + (i & 7) + ((lane >> 4) << 3) + ((i >> 3) << 4);
    int n    = nt * 16 + (lane & 15);
    Bp[t] = (__bf16)B[(size_t)kb * N + n];
}

// ---------------------------------------------------------------------------
// 5) bf16 WMMA GEMM, one wave per 16x64 C strip (4 N-tiles share one A frag).
//    A (16x32 bf16, ISA 7.12.2): lane l15 = row; per lane two contiguous runs:
//      K in [k0+8*lhalf, +8) and [k0+16+8*lhalf, +8)  -> 2x b128 loads
//    B pre-packed in fragment order -> contiguous 32B/lane load per tile.
//    C/D (16x16 f32): VGPR r = row r + 8*lhalf, lane l15 = column.
// ---------------------------------------------------------------------------
template <int ACT, int OUTBF>   // ACT: 0 none, 1 leaky_relu(0.01); OUTBF: bf16/f32 store
__global__ void k_wmma_gemm(const __bf16* __restrict__ A, int lda,
                            const __bf16* __restrict__ Bp,
                            const float* __restrict__ bias,
                            void* __restrict__ Cout, int ldc,
                            int M, int N, int K) {
    int lane    = threadIdx.x & 31;
    int wib     = threadIdx.x >> 5;
    int strip   = blockIdx.x * (blockDim.x >> 5) + wib;
    int nstrips = N >> 6;                 // strips of 64 columns
    int mt      = strip / nstrips;
    int nt0     = (strip - mt * nstrips) << 2;
    if (mt * 16 >= M) return;             // wave-uniform
    int ntn   = N >> 4;
    int lhalf = lane >> 4;
    int l15   = lane & 15;

    const __bf16* Arow = A + (size_t)(mt * 16 + l15) * lda + (lhalf << 3);
    const __bf16* bp   = Bp + ((size_t)nt0 * 32 + lane) * 16;
    const size_t  bstride = (size_t)ntn * 512;   // packed elements per K-step

    v8f c0 = {}, c1 = {}, c2 = {}, c3 = {};
    int ksteps = K >> 5;
    for (int kt = 0; kt < ksteps; ++kt) {
        int k0 = kt << 5;
        v8bf alo = *(const v8bf*)(Arow + k0);
        v8bf ahi = *(const v8bf*)(Arow + k0 + 16);
        __builtin_prefetch(Arow + k0 + 32, 0, 3);
        v16bf aF = __builtin_shufflevector(alo, ahi,
                     0, 1, 2, 3, 4, 5, 6, 7, 8, 9, 10, 11, 12, 13, 14, 15);
        v16bf b0 = *(const v16bf*)(bp);
        v16bf b1 = *(const v16bf*)(bp + 512);
        v16bf b2 = *(const v16bf*)(bp + 1024);
        v16bf b3 = *(const v16bf*)(bp + 1536);
        bp += bstride;
        c0 = __builtin_amdgcn_wmma_f32_16x16x32_bf16(false, aF, false, b0, (short)0, c0, false, false);
        c1 = __builtin_amdgcn_wmma_f32_16x16x32_bf16(false, aF, false, b1, (short)0, c1, false, false);
        c2 = __builtin_amdgcn_wmma_f32_16x16x32_bf16(false, aF, false, b2, (short)0, c2, false, false);
        c3 = __builtin_amdgcn_wmma_f32_16x16x32_bf16(false, aF, false, b3, (short)0, c3, false, false);
    }

    int mbase = mt * 16 + (lhalf << 3);
    v8f accs[4] = {c0, c1, c2, c3};
    #pragma unroll
    for (int tIdx = 0; tIdx < 4; ++tIdx) {
        int n = (nt0 + tIdx) * 16 + l15;
        float bv = bias ? bias[n] : 0.f;
        #pragma unroll
        for (int r = 0; r < 8; ++r) {
            float v = accs[tIdx][r] + bv;
            if (ACT == 1) v = (v > 0.f) ? v : 0.01f * v;
            size_t o = (size_t)(mbase + r) * ldc + n;
            if (OUTBF) ((__bf16*)Cout)[o] = (__bf16)v;
            else       ((float*)Cout)[o] = v;
        }
    }
}

// ---------------------------------------------------------------------------
// 6) Sparse normalized propagate: acc[u,:] += edge_w * d_c[v] * Y[v,:]
// ---------------------------------------------------------------------------
__global__ void k_scatter_prop(const int* __restrict__ u_idx, const int* __restrict__ v_idx,
                               const float* __restrict__ edge_w, const float* __restrict__ c_sum,
                               const float* __restrict__ Yf, float* __restrict__ acc) {
    int t = blockIdx.x * blockDim.x + threadIdx.x;
    const int chunks = HV_ / 4;
    if (t >= 2 * NE_ * chunks) return;
    int j  = t >> 6;
    int h4 = (t & 63) << 2;
    int e  = (j < NE_) ? j : j - NE_;
    int u  = (j < NE_) ? u_idx[e] : v_idx[e];
    int v  = (j < NE_) ? v_idx[e] : u_idx[e];
    float w = edge_w[j] * rsqrtf(fmaxf(c_sum[v], EPS_));
    float4 y = *(const float4*)(Yf + (size_t)v * HV_ + h4);
    float* dst = acc + (size_t)u * HV_ + h4;
    atomicAdd(dst + 0, w * y.x);
    atomicAdd(dst + 1, w * y.y);
    atomicAdd(dst + 2, w * y.z);
    atomicAdd(dst + 3, w * y.w);
}

// 7) hidden = tanh(d_r[v] * acc) -> second half of concat buffer (bf16)
__global__ void k_hidden(const float* __restrict__ acc, const float* __restrict__ r_sum,
                         __bf16* __restrict__ h_cat) {
    int t = blockIdx.x * blockDim.x + threadIdx.x;
    if (t >= NV_ * HV_) return;
    int v = t >> 8, h = t & (HV_ - 1);
    float dr = rsqrtf(fmaxf(r_sum[v], EPS_));
    h_cat[(size_t)v * (2 * HV_) + HV_ + h] = (__bf16)tanhf(dr * acc[t]);
}

// ---------------------------------------------------------------------------
// Host-side orchestration
// ---------------------------------------------------------------------------
extern "C" void kernel_launch(void* const* d_in, const int* in_sizes, int n_in,
                              void* d_out, int out_size, void* d_ws, size_t ws_size,
                              hipStream_t stream) {
    (void)in_sizes; (void)n_in; (void)out_size; (void)ws_size;

    const float* hv_ftr = (const float*)d_in[0];   // [V, HV]
    const float* he_ftr = (const float*)d_in[1];   // [E, HE]
    const float* vew1   = (const float*)d_in[2];   // [V, E] one-hot
    const float* vew2   = (const float*)d_in[3];   // [V, E] one-hot
    const float* w_att  = (const float*)d_in[4];   // [2HV+HE, 1]
    const float* b_att  = (const float*)d_in[5];   // [1]
    const float* w_gcn  = (const float*)d_in[6];   // [HV, HV]
    const float* b_gcn  = (const float*)d_in[7];   // [HV]
    const float* w1     = (const float*)d_in[8];   // [2HV, HV]
    const float* b1     = (const float*)d_in[9];   // [HV]
    const float* w2     = (const float*)d_in[10];  // [HV, OUT]
    const float* b2     = (const float*)d_in[11];  // [OUT]
    float*       out    = (float*)d_out;           // [V, OUT] f32

    // -------- workspace carve-up (zeroed region first) --------
    size_t off = 0;
    auto carve = [&](size_t bytes) -> void* {
        void* p = (char*)d_ws + off;
        off += (bytes + 255) & ~(size_t)255;
        return p;
    };
    float* r_sum  = (float*)carve(NV_ * sizeof(float));               // zeroed
    float* c_sum  = (float*)carve(NV_ * sizeof(float));               // zeroed
    float* acc    = (float*)carve((size_t)NV_ * HV_ * sizeof(float)); // zeroed
    size_t zero_bytes = off;
    int*    u_idx  = (int*)carve(NE_ * sizeof(int));
    int*    v_idx  = (int*)carve(NE_ * sizeof(int));
    float*  a_vec  = (float*)carve(NV_ * sizeof(float));
    float*  c_vec  = (float*)carve(NV_ * sizeof(float));
    float*  be_vec = (float*)carve(NE_ * sizeof(float));
    float*  edge_w = (float*)carve(2 * NE_ * sizeof(float));
    __bf16* h_cat  = (__bf16*)carve((size_t)NV_ * 2 * HV_ * sizeof(__bf16));
    __bf16* wgcnp  = (__bf16*)carve((size_t)HV_ * HV_ * sizeof(__bf16));
    __bf16* w1p    = (__bf16*)carve((size_t)2 * HV_ * HV_ * sizeof(__bf16));
    __bf16* w2p    = (__bf16*)carve((size_t)HV_ * OUT_ * sizeof(__bf16));
    float*  Yf     = (float*)carve((size_t)NV_ * HV_ * sizeof(float));
    __bf16* h1bf   = (__bf16*)carve((size_t)NV_ * HV_ * sizeof(__bf16));

    hipMemsetAsync(d_ws, 0, zero_bytes, stream);

    // -------- 1) recover edge indices (bandwidth-bound 256MB scan) --------
    {
        int total4 = NV_ * NE_ / 4;
        int blocks = (total4 + 255) / 256;
        k_extract_idx<<<blocks, 256, 0, stream>>>((const float4*)vew1, u_idx, total4);
        k_extract_idx<<<blocks, 256, 0, stream>>>((const float4*)vew2, v_idx, total4);
    }

    // -------- 2) attention projections --------
    k_att_hv<<<NV_ / 4, 128, 0, stream>>>(hv_ftr, w_att, a_vec, c_vec);
    k_att_he<<<NE_ / 4, 128, 0, stream>>>(he_ftr, w_att, be_vec);

    // -------- bf16 conversions + weight packing (fragment order) --------
    k_hv2cat<<<(NV_ * HV_) / 256, 256, 0, stream>>>(hv_ftr, h_cat);
    k_packB<<<(HV_ * HV_) / 256, 256, 0, stream>>>(w_gcn, wgcnp, HV_, HV_);
    k_packB<<<(2 * HV_ * HV_) / 256, 256, 0, stream>>>(w1, w1p, 2 * HV_, HV_);
    k_packB<<<(HV_ * OUT_) / 256, 256, 0, stream>>>(w2, w2p, HV_, OUT_);

    // -------- 3) edge weights + degree sums --------
    k_edge<<<(2 * NE_) / 256, 256, 0, stream>>>(u_idx, v_idx, a_vec, c_vec, be_vec,
                                                b_att, edge_w, r_sum, c_sum);

    // -------- 4) Y = hv @ w_gcn + b_gcn  (WMMA, f32 out) --------
    {
        int strips = (NV_ / 16) * (HV_ / 64);
        k_wmma_gemm<0, 0><<<strips / 4, 128, 0, stream>>>(
            h_cat, 2 * HV_, wgcnp, b_gcn, Yf, HV_, NV_, HV_, HV_);
    }

    // -------- 5) sparse normalized propagate --------
    k_scatter_prop<<<(2 * NE_ * (HV_ / 4)) / 256, 256, 0, stream>>>(
        u_idx, v_idx, edge_w, c_sum, Yf, acc);

    // -------- 6) hidden = tanh(d_r * acc) into concat buffer --------
    k_hidden<<<(NV_ * HV_) / 256, 256, 0, stream>>>(acc, r_sum, h_cat);

    // -------- 7) h1 = leaky_relu([hv, hidden] @ w1 + b1)  (WMMA, bf16 out) --------
    {
        int strips = (NV_ / 16) * (HV_ / 64);
        k_wmma_gemm<1, 1><<<strips / 4, 128, 0, stream>>>(
            h_cat, 2 * HV_, w1p, b1, h1bf, HV_, NV_, HV_, 2 * HV_);
    }

    // -------- 8) out = h1 @ w2 + b2  (WMMA, f32 out) --------
    {
        int strips = (NV_ / 16) * (OUT_ / 64);
        k_wmma_gemm<0, 0><<<strips / 4, 128, 0, stream>>>(
            h1bf, HV_, w2p, b2, out, OUT_, NV_, OUT_, HV_);
    }
}
